// SolarRingLayer_12610023981238
// MI455X (gfx1250) — matrix operand order repaired and checked by reading, not compiled, not run on hardware
//
#include <hip/hip_runtime.h>

typedef __bf16 bf16_t;
typedef __attribute__((ext_vector_type(16))) __bf16 v16bf;
typedef __attribute__((ext_vector_type(8)))  __bf16 v8bf;
typedef __attribute__((ext_vector_type(8)))  float  v8f;

constexpr int DIM   = 1024;
constexpr int NROWS = 65536;
constexpr int RINGS = 4096;
constexpr int LDX   = DIM + 8;     // LDS row stride (bf16 elems), 16B-aligned, bank-skewed
constexpr int MT    = 128;         // rows per workgroup (8 waves x 16)

__device__ __forceinline__ float sigmoidf_(float v) { return 1.0f / (1.0f + __expf(-v)); }

__device__ __forceinline__ v16bf make_v16(v8bf lo, v8bf hi) {
  return __builtin_shufflevector(lo, hi, 0,1,2,3,4,5,6,7,8,9,10,11,12,13,14,15);
}

// A-matrix 16x32 bf16 fragment (M across lanes 0..15; K=0..15 lanes 0-15, K=16..31 lanes 16-31)
__device__ __forceinline__ v16bf load_a_frag(const bf16_t* base, int stride, int row0, int k0) {
  int lane = threadIdx.x & 31;
  const bf16_t* p = base + (size_t)(row0 + (lane & 15)) * stride + k0 + ((lane >> 4) << 4);
  v8bf lo = *(const v8bf*)p;
  v8bf hi = *(const v8bf*)(p + 8);
  return make_v16(lo, hi);
}

// B-matrix 32x16 bf16 fragment (N across lanes 0..15; K=0..15 lanes 0-15, K=16..31 lanes 16-31)
__device__ __forceinline__ v16bf load_b_frag(const bf16_t* base, int stride, int n0, int k0) {
  int lane = threadIdx.x & 31;
  const bf16_t* p = base + (size_t)(n0 + (lane & 15)) * stride + k0 + ((lane >> 4) << 4);
  return *(const v16bf*)p;
}

__device__ __forceinline__ v8f wmma_bf16(v16bf a, v16bf b, v8f c) {
  return __builtin_amdgcn_wmma_f32_16x16x32_bf16(false, a, false, b, (short)0, c, false, false);
}

// stage 8 fp32 -> 8 bf16 into LDS
__device__ __forceinline__ void cvt_store8(bf16_t* dstp, const float* src) {
  float4 a = *(const float4*)(src);
  float4 b = *(const float4*)(src + 4);
  v8bf t;
  t[0] = (bf16_t)a.x; t[1] = (bf16_t)a.y; t[2] = (bf16_t)a.z; t[3] = (bf16_t)a.w;
  t[4] = (bf16_t)b.x; t[5] = (bf16_t)b.y; t[6] = (bf16_t)b.z; t[7] = (bf16_t)b.w;
  *(v8bf*)(dstp) = t;
}

// ---------------- setup kernels ----------------

__global__ void zero_cnt_kernel(int* p) {
  if (threadIdx.x < 8) p[threadIdx.x] = 0;
}

__global__ void convert5_kernel(const float* __restrict__ s0, const float* __restrict__ s1,
                                const float* __restrict__ s2, const float* __restrict__ s3,
                                const float* __restrict__ s4, bf16_t* __restrict__ dst) {
  const unsigned total = 5u << 20;
  for (unsigned i = blockIdx.x * blockDim.x + threadIdx.x; i < total;
       i += gridDim.x * blockDim.x) {
    unsigned w = i >> 20, r = i & 0xFFFFFu;
    const float* s = (w == 0) ? s0 : (w == 1) ? s1 : (w == 2) ? s2 : (w == 3) ? s3 : s4;
    dst[i] = (bf16_t)s[r];
  }
}

// pack [W_role(9) ; W_spawn(1) ; W_vg(1) ; zeros(5)] into 16x1024 bf16
__global__ void build_small_kernel(const float* __restrict__ Wr, const float* __restrict__ Wsp,
                                   const float* __restrict__ Wv, bf16_t* __restrict__ dst) {
  for (int i = blockIdx.x * blockDim.x + threadIdx.x; i < 16 * DIM;
       i += gridDim.x * blockDim.x) {
    int r = i >> 10, col = i & (DIM - 1);
    float v = 0.0f;
    if (r < 9)       v = Wr[r * DIM + col];
    else if (r == 9) v = Wsp[col];
    else if (r == 10) v = Wv[col];
    dst[i] = (bf16_t)v;
  }
}

// cats: 0=subj(role1) 1=obj(role2) 2=verb(role3) 3=rot(other) 4=conj(role5)
__global__ void classify_kernel(const int* __restrict__ roles, int* cnt, int* lists) {
  for (int i = blockIdx.x * blockDim.x + threadIdx.x; i < NROWS;
       i += gridDim.x * blockDim.x) {
    int r = roles[i];
    int c = (r == 1) ? 0 : (r == 2) ? 1 : (r == 3) ? 2 : (r == 5) ? 4 : 3;
    int p = atomicAdd(&cnt[c], 1);
    lists[c * NROWS + p] = i;
  }
}

// ---------------- main fused kernel: og-gate GEMM + small GEMM + y + LN stats ----------------

__global__ void __launch_bounds__(256) main_kernel(
    const float* __restrict__ x, const bf16_t* __restrict__ Wog,
    const float* __restrict__ b_og, const bf16_t* __restrict__ Wsm,
    const float* __restrict__ b_role, const float* __restrict__ b_spawn,
    const float* __restrict__ b_vg,
    float* __restrict__ out_x, float* __restrict__ out_role,
    float* __restrict__ out_spawn, float* __restrict__ ws_vg,
    float* __restrict__ ws_sum, float* __restrict__ ws_sq) {
  extern __shared__ char smem[];
  bf16_t* xs = (bf16_t*)smem;                  // MT x LDX bf16
  const int tid = threadIdx.x;
  const int rowBase = blockIdx.x * MT;

  // stage MT rows of x -> LDS bf16 (2 threads per row)
  {
    int r = tid >> 1, half = tid & 1;
    const float* src = x + (size_t)(rowBase + r) * DIM + half * 512;
    bf16_t* dstp = xs + r * LDX + half * 512;
    for (int c = 0; c < 512; c += 8) cvt_store8(dstp + c, src + c);
  }
  __syncthreads();

  const int wave = tid >> 5, lane = tid & 31;
  const int m0 = wave * 16;
  const int nsel = lane & 15;
  const int mb = m0 + ((lane >> 4) << 3);      // base row (local) of this lane's 8 C values

  float rs[8], rq[8];
#pragma unroll
  for (int i = 0; i < 8; ++i) { rs[i] = 0.0f; rq[i] = 0.0f; }

  // epilogue for one 16-col tile of the gate GEMM
  auto epilogue = [&](int j0, v8f acc) {
    const int n = j0 + nsel;
    const float bo = b_og[n];
#pragma unroll
    for (int i = 0; i < 8; ++i) {
      const size_t row = (size_t)(rowBase + mb + i);
      float g = sigmoidf_(acc[i] + bo);
      float xv = x[row * DIM + n];
      float y = xv * (1.0f + g);
      out_x[row * DIM + n] = y;       // un-normalized; LN pass fixes it up
      rs[i] += y;
      rq[i] += y * y;
    }
  };

  // two j-tiles per pass: 2 independent WMMA chains, A fragments reused
  for (int jt = 0; jt < 64; jt += 2) {
    const int j0 = jt * 16, j1 = j0 + 16;
    v8f acc0 = {}, acc1 = {};
#pragma unroll 4
    for (int k = 0; k < DIM; k += 32) {
      v16bf a  = load_a_frag(xs, LDX, m0, k);
      v16bf b0 = load_b_frag(Wog, DIM, j0, k);
      v16bf b1 = load_b_frag(Wog, DIM, j1, k);
      acc0 = wmma_bf16(a, b0, acc0);
      acc1 = wmma_bf16(a, b1, acc1);
    }
    epilogue(j0, acc0);
    epilogue(j1, acc1);
  }

  // reduce row sums across the 16 lanes of each half (wave32)
#pragma unroll
  for (int i = 0; i < 8; ++i) {
    for (int m = 1; m < 16; m <<= 1) {
      rs[i] += __shfl_xor(rs[i], m, 32);
      rq[i] += __shfl_xor(rq[i], m, 32);
    }
  }
  if (nsel == 0) {
#pragma unroll
    for (int i = 0; i < 8; ++i) {
      int row = rowBase + mb + i;
      ws_sum[row] = rs[i];
      ws_sq[row]  = rq[i];
    }
  }

  // small GEMM: role logits (9) + spawn (1) + verb gate (1) in one j-tile
  {
    v8f acc = {};
#pragma unroll 4
    for (int k = 0; k < DIM; k += 32) {
      v16bf a = load_a_frag(xs, LDX, m0, k);
      v16bf b = load_b_frag(Wsm, DIM, 0, k);
      acc = wmma_bf16(a, b, acc);
    }
    const int n = nsel;
#pragma unroll
    for (int i = 0; i < 8; ++i) {
      const size_t row = (size_t)(rowBase + mb + i);
      float v = acc[i];
      if (n < 9)        out_role[row * 9 + n] = v + b_role[n];
      else if (n == 9)  out_spawn[row] = v + b_spawn[0];
      else if (n == 10) ws_vg[row] = sigmoidf_(v + b_vg[0]);
    }
  }
}

// ---------------- LN finalize ----------------

__global__ void ln_kernel(float* __restrict__ out_x, const float* __restrict__ s,
                          const float* __restrict__ q, const float* __restrict__ g,
                          const float* __restrict__ b) {
  const size_t total = (size_t)NROWS * DIM / 4;
  for (size_t i = (size_t)blockIdx.x * blockDim.x + threadIdx.x; i < total;
       i += (size_t)gridDim.x * blockDim.x) {
    size_t e = i * 4;
    int row = (int)(e >> 10);
    int col = (int)(e & (DIM - 1));
    float mu  = s[row] * (1.0f / 1024.0f);
    float var = q[row] * (1.0f / 1024.0f) - mu * mu;
    float inv = rsqrtf(var + 1e-5f);
    float4 y  = *(float4*)(out_x + e);
    float4 gg = *(const float4*)(g + col);
    float4 bb = *(const float4*)(b + col);
    y.x = (y.x - mu) * inv * gg.x + bb.x;
    y.y = (y.y - mu) * inv * gg.y + bb.y;
    y.z = (y.z - mu) * inv * gg.z + bb.z;
    y.w = (y.w - mu) * inv * gg.w + bb.w;
    *(float4*)(out_x + e) = y;
  }
}

// ---------------- merged category GEMM + scatter-add ----------------

__global__ void __launch_bounds__(256) cat_kernel(
    const float* __restrict__ x, const int* __restrict__ cnt, const int* __restrict__ lists,
    const int* __restrict__ slots, const float* __restrict__ ws_vg,
    const bf16_t* __restrict__ w0, const bf16_t* __restrict__ w1,
    const bf16_t* __restrict__ w2, const bf16_t* __restrict__ w3,
    const float* __restrict__ b0, const float* __restrict__ b1,
    const float* __restrict__ b2, const float* __restrict__ b3,
    float* __restrict__ out_mem) {
  extern __shared__ char smem[];
  bf16_t* xs     = (bf16_t*)smem;
  int*    rowIds = (int*)(smem + (size_t)MT * LDX * sizeof(bf16_t));
  int*    slotA  = rowIds + MT;
  float*  vgA    = (float*)(slotA + MT);

  // map block -> (category, tile)
  int counts[4] = {cnt[0], cnt[1], cnt[2], cnt[3]};
  int c = -1, t = blockIdx.x;
  for (int i = 0; i < 4; ++i) {
    int T = (counts[i] + MT - 1) >> 7;
    if (t < T) { c = i; break; }
    t -= T;
  }
  if (c < 0) return;
  const int ccount = counts[c];
  const int base = t * MT;
  const bf16_t* W   = (c == 0) ? w0 : (c == 1) ? w1 : (c == 2) ? w2 : w3;
  const float* bias = (c == 0) ? b0 : (c == 1) ? b1 : (c == 2) ? b2 : b3;

  const int tid = threadIdx.x;
  if (tid < MT) {
    int pos = base + tid;
    int id = (pos < ccount) ? lists[c * NROWS + pos] : -1;
    rowIds[tid] = id;
    slotA[tid]  = (id >= 0) ? slots[id] : 0;
    vgA[tid]    = (id >= 0 && c == 2) ? ws_vg[id] : 1.0f;
  }
  __syncthreads();

  // gather + stage rows -> LDS bf16
  {
    int r = tid >> 1, half = tid & 1;
    int id = rowIds[r];
    bf16_t* dstp = xs + r * LDX + half * 512;
    if (id >= 0) {
      const float* src = x + (size_t)id * DIM + half * 512;
      for (int cc = 0; cc < 512; cc += 8) cvt_store8(dstp + cc, src + cc);
    } else {
      for (int cc = 0; cc < 512; cc += 8) {
        v8bf z = {};
        *(v8bf*)(dstp + cc) = z;
      }
    }
  }
  __syncthreads();

  const int wave = tid >> 5, lane = tid & 31;
  const int m0 = wave * 16;
  const int nsel = lane & 15;
  const int mb = m0 + ((lane >> 4) << 3);

  bool  valR[8];
  int   slotR[8];
  float vgR[8];
#pragma unroll
  for (int i = 0; i < 8; ++i) {
    int rloc = mb + i;
    valR[i]  = rowIds[rloc] >= 0;
    slotR[i] = slotA[rloc];
    vgR[i]   = vgA[rloc];
  }

  auto epilogue = [&](int j0, v8f acc) {
    const int n = j0 + nsel;
    const float bn = bias[n];
#pragma unroll
    for (int i = 0; i < 8; ++i) {
      if (valR[i]) {
        float v = (acc[i] + bn) * vgR[i];
        atomicAdd(&out_mem[(size_t)slotR[i] * DIM + n], v);
      }
    }
  };

  for (int jt = 0; jt < 64; jt += 2) {
    const int j0 = jt * 16, j1 = j0 + 16;
    v8f acc0 = {}, acc1 = {};
#pragma unroll 4
    for (int k = 0; k < DIM; k += 32) {
      v16bf a  = load_a_frag(xs, LDX, m0, k);
      v16bf b0 = load_b_frag(W, DIM, j0, k);
      v16bf b1 = load_b_frag(W, DIM, j1, k);
      acc0 = wmma_bf16(a, b0, acc0);
      acc1 = wmma_bf16(a, b1, acc1);
    }
    epilogue(j0, acc0);
    epilogue(j1, acc1);
  }
}

// ---------------- conj rows: write_vec = x, direct scatter ----------------

__global__ void conj_kernel(const float* __restrict__ x, const int* __restrict__ cnt,
                            const int* __restrict__ lists, const int* __restrict__ slots,
                            float* __restrict__ out_mem) {
  int total = cnt[4];
  for (int r = blockIdx.x; r < total; r += gridDim.x) {
    int id = lists[4 * NROWS + r];
    const float* src = x + (size_t)id * DIM;
    float* dst = out_mem + (size_t)slots[id] * DIM;
    for (int c = threadIdx.x * 4; c < DIM; c += blockDim.x * 4) {
      float4 v = *(const float4*)(src + c);
      atomicAdd(dst + c + 0, v.x);
      atomicAdd(dst + c + 1, v.y);
      atomicAdd(dst + c + 2, v.z);
      atomicAdd(dst + c + 3, v.w);
    }
  }
}

// ---------------- launch ----------------

extern "C" void kernel_launch(void* const* d_in, const int* in_sizes, int n_in,
                              void* d_out, int out_size, void* d_ws, size_t ws_size,
                              hipStream_t stream) {
  const float* x       = (const float*)d_in[0];
  const int*   roles   = (const int*)d_in[1];
  const int*   slots   = (const int*)d_in[2];
  const float* memory  = (const float*)d_in[3];
  const float* W_role  = (const float*)d_in[4];
  const float* b_role  = (const float*)d_in[5];
  const float* W_spawn = (const float*)d_in[6];
  const float* b_spawn = (const float*)d_in[7];
  const float* W_subj  = (const float*)d_in[8];
  const float* b_subj  = (const float*)d_in[9];
  const float* W_obj   = (const float*)d_in[10];
  const float* b_obj   = (const float*)d_in[11];
  const float* W_vg    = (const float*)d_in[12];
  const float* b_vg    = (const float*)d_in[13];
  const float* W_vc    = (const float*)d_in[14];
  const float* b_vc    = (const float*)d_in[15];
  const float* W_rot   = (const float*)d_in[16];
  const float* b_rot   = (const float*)d_in[17];
  const float* W_og    = (const float*)d_in[18];
  const float* b_og    = (const float*)d_in[19];
  const float* ln_g    = (const float*)d_in[20];
  const float* ln_b    = (const float*)d_in[21];

  float* out_x     = (float*)d_out;
  float* out_role  = out_x + (size_t)NROWS * DIM;
  float* out_spawn = out_role + (size_t)NROWS * 9;
  float* out_mem   = out_spawn + NROWS;

  // workspace layout
  char* ws = (char*)d_ws;
  bf16_t* wBig   = (bf16_t*)ws;                        // 5 * 1M bf16 = 10 MB
  bf16_t* wSmall = wBig + (size_t)5 * 1024 * 1024;     // 16 * 1024 bf16
  size_t off = ((size_t)5 * 1024 * 1024 + 16 * 1024) * sizeof(bf16_t);
  off = (off + 255) & ~(size_t)255;
  int*   cnt    = (int*)(ws + off);                    // 8 counters (use 64-int slot)
  int*   lists  = cnt + 64;                            // 5 * NROWS ints
  float* ws_vg  = (float*)(lists + (size_t)5 * NROWS); // NROWS
  float* ws_sum = ws_vg + NROWS;                       // NROWS
  float* ws_sq  = ws_sum + NROWS;                      // NROWS

  // 1) memory_new starts as memory
  hipMemcpyAsync(out_mem, memory, (size_t)RINGS * DIM * sizeof(float),
                 hipMemcpyDeviceToDevice, stream);

  // 2) setup
  zero_cnt_kernel<<<1, 32, 0, stream>>>(cnt);
  convert5_kernel<<<4096, 256, 0, stream>>>(W_subj, W_obj, W_vc, W_rot, W_og, wBig);
  build_small_kernel<<<64, 256, 0, stream>>>(W_role, W_spawn, W_vg, wSmall);
  classify_kernel<<<256, 256, 0, stream>>>(roles, cnt, lists);

  // 3) main fused GEMM (og gate + role/spawn/vg) + y + LN stats
  size_t ldsMain = (size_t)MT * LDX * sizeof(bf16_t);  // 264192 B (<320KB/WGP)
  main_kernel<<<NROWS / MT, 256, ldsMain, stream>>>(
      x, wBig + (size_t)4 * 1024 * 1024, b_og, wSmall, b_role, b_spawn, b_vg,
      out_x, out_role, out_spawn, ws_vg, ws_sum, ws_sq);

  // 4) LN finalize
  ln_kernel<<<4096, 256, 0, stream>>>(out_x, ws_sum, ws_sq, ln_g, ln_b);

  // 5) per-category GEMM + scatter (subj/obj/verb/rot merged into one launch)
  size_t ldsCat = ldsMain + MT * (sizeof(int) * 2 + sizeof(float));
  cat_kernel<<<1024, 256, ldsCat, stream>>>(
      x, cnt, lists, slots, ws_vg,
      wBig + (size_t)0 * 1024 * 1024, wBig + (size_t)1 * 1024 * 1024,
      wBig + (size_t)2 * 1024 * 1024, wBig + (size_t)3 * 1024 * 1024,
      b_subj, b_obj, b_vc, b_rot, out_mem);

  // 6) conj rows scatter x directly
  conj_kernel<<<1024, 256, 0, stream>>>(x, cnt, lists, slots, out_mem);
}
